// NerveNetGNN_18872086298900
// MI455X (gfx1250) — compile-verified, matrix operand order checked by reading, not run on hardware
//
#include <hip/hip_runtime.h>
#include <hip/hip_bf16.h>
#include <math.h>

// ---------------------------------------------------------------------------
// Types for CDNA5 WMMA (wave32): D(16x16 f32) = A(16x32 bf16) x B(32x16 bf16) + C
// ---------------------------------------------------------------------------
typedef __bf16 bf16_t;
typedef bf16_t v16bf __attribute__((ext_vector_type(16)));
typedef float  v8f   __attribute__((ext_vector_type(8)));

#define B_SZ   2048
#define NND    64
#define D_IN   128
#define D_H    256
#define D_OUT  512
#define OBS_N  416
#define KC     64     // K elements staged per pipeline stage in heads GEMM

#if __has_builtin(__builtin_amdgcn_tensor_load_to_lds)
#define HAVE_TDM 1
#else
#define HAVE_TDM 0
#endif

// ---------------------------------------------------------------------------
// Fragment loader (CDNA5 ISA 7.12.2, 16-bit 16x32 operand layout).
// p must already point at the lane's row (base + row*ld); rows are K-contiguous.
// dword j holds K-pair:  j<4: k=kbase+2j+8*half ; j>=4: k=kbase+16+2(j-4)+8*half
// ---------------------------------------------------------------------------
__device__ __forceinline__ v16bf ldfrag(const bf16_t* p, int kbase, int half) {
  union { v16bf v; unsigned u[8]; } r;
#pragma unroll
  for (int j = 0; j < 8; ++j) {
    int k = kbase + ((j < 4) ? (2 * j) : (8 + 2 * j)) + 8 * half;
    r.u[j] = *(const unsigned*)(p + k);
  }
  return r.v;
}

__device__ __forceinline__ v8f wmma_bf16(v16bf a, v16bf b, v8f c) {
  return __builtin_amdgcn_wmma_f32_16x16x32_bf16(false, a, false, b, (short)0, c,
                                                 false, false);
}

// ---------------------------------------------------------------------------
// CDNA5 async global->LDS copy (ASYNCcnt) and TDM tensor load (TENSORcnt).
// ---------------------------------------------------------------------------
__device__ __forceinline__ unsigned ldsaddr(const void* p) {
  return (unsigned)(unsigned long long)(__attribute__((address_space(3))) const void*)p;
}

__device__ __forceinline__ void async_copy_b128(unsigned lds_byte_addr,
                                                const void* gsrc) {
  asm volatile("global_load_async_to_lds_b128 %0, %1, off"
               :: "v"(lds_byte_addr), "v"(gsrc)
               : "memory");
}

template <int N>
__device__ __forceinline__ void wait_async_le() {
#if __has_builtin(__builtin_amdgcn_s_wait_asynccnt)
  __builtin_amdgcn_s_wait_asynccnt((short)N);
#else
  asm volatile("s_wait_asynccnt %0" :: "i"(N) : "memory");
#endif
}

#if HAVE_TDM
// 2D tile load via Tensor Data Mover: 'rows' rows of KC bf16 elements, source
// row pitch = K elements, packed contiguously into LDS (rows * KC * 2 bytes).
// D# per CDNA5 ISA 8.3/8.4 (group0: count/lds/global/type, group1: dims+stride).
// This toolchain declares the 6-arg builtin:
//   (uint32x4 g0, int32x8 g1, int32x4 g2, int32x4 g3, int32x8 g4, int cpol)
__device__ __forceinline__ void tdm_load_tile(unsigned lds_addr,
                                              unsigned long long gaddr,
                                              unsigned K_elems, unsigned rows) {
  typedef unsigned u32x4 __attribute__((ext_vector_type(4)));
  typedef int i32x8 __attribute__((ext_vector_type(8)));
  typedef int i32x4 __attribute__((ext_vector_type(4)));
  u32x4 g0 = {
      1u,                                                  // count=1, user mode
      lds_addr,                                            // lds_addr[31:0]
      (unsigned)gaddr,                                     // global_addr[31:0]
      ((unsigned)(gaddr >> 32) & 0x01FFFFFFu) | (2u << 30) // addr[56:32] | type=2
  };
  i32x8 g1 = {
      (int)(1u << 16),              // workgroup_mask=0, data_size=1 (2 bytes)
      (int)((KC & 0xFFFFu) << 16),  // tensor_dim0[15:0] = KC  (bits 63:48)
      (int)(rows << 16),            // dim0 hi=0 ; tensor_dim1[15:0]=rows (95:80)
      (int)((unsigned)KC << 16),    // dim1 hi=0 ; tile_dim0=KC (127:112)
      (int)rows,                    // tile_dim1=rows (143:128), tile_dim2=0
      (int)K_elems,                 // tensor_dim0_stride[31:0] (elements)
      0, 0                          // stride hi / dim1_stride = 0
  };
  i32x4 z4 = {0, 0, 0, 0};
  i32x8 z8 = {0, 0, 0, 0, 0, 0, 0, 0};
  __builtin_amdgcn_tensor_load_to_lds(g0, g1, z4, z4, z8, 0);
}

template <int N>
__device__ __forceinline__ void wait_tensor_le() {
  __builtin_amdgcn_s_wait_tensorcnt((short)N);
}
#endif

// ---------------------------------------------------------------------------
// Kernel 1: graph prep. deg -> dinv -> dense normalized adjacency (64x64) bf16.
// ---------------------------------------------------------------------------
__global__ void prep_kernel(const int* __restrict__ esrc,
                            const int* __restrict__ edst,
                            bf16_t* __restrict__ adj_bf, int nedges) {
  __shared__ float deg[NND];
  __shared__ float dinv[NND];
  __shared__ float adjf[NND * NND];
  int t = threadIdx.x;
  if (t < NND) deg[t] = 0.f;
  for (int i = t; i < NND * NND; i += blockDim.x) adjf[i] = 0.f;
  __syncthreads();
  if (t < nedges) atomicAdd(&deg[edst[t]], 1.0f);
  __syncthreads();
  if (t < NND) dinv[t] = (deg[t] > 0.f) ? rsqrtf(deg[t]) : 0.f;
  __syncthreads();
  if (t < nedges) {
    int s = esrc[t], d = edst[t];
    atomicAdd(&adjf[d * NND + s], dinv[s] * dinv[d]);
  }
  __syncthreads();
  for (int i = t; i < NND * NND; i += blockDim.x) adj_bf[i] = (bf16_t)adjf[i];
}

// ---------------------------------------------------------------------------
// Kernel 2: weight convert+transpose.  src (K,N) f32 row-major -> Wt[n*K+k] bf16
// ---------------------------------------------------------------------------
__global__ void wcvt_kernel(const float* __restrict__ W, bf16_t* __restrict__ Wt,
                            int K, int N) {
  int idx = blockIdx.x * blockDim.x + threadIdx.x;
  if (idx < K * N) {
    int k = idx / N, n = idx % N;
    Wt[(size_t)n * K + k] = (bf16_t)W[idx];
  }
}

// ---------------------------------------------------------------------------
// Kernel 3: gather + per-group input embedding + tanh -> emb_bf (B,64,128)
// ---------------------------------------------------------------------------
__global__ void embed_kernel(const float* __restrict__ obs,
                             const int* __restrict__ obs_idx,
                             const float* __restrict__ W0, const float* __restrict__ b0,
                             const float* __restrict__ W1, const float* __restrict__ b1,
                             const float* __restrict__ W2, const float* __restrict__ b2,
                             const float* __restrict__ W3, const float* __restrict__ b3,
                             bf16_t* __restrict__ emb_bf) {
  int b = blockIdx.x, n = blockIdx.y, k = threadIdx.x;
  int g = n >> 4;
  const float* W; const float* bias; int d;
  switch (g) {
    case 0:  W = W0; bias = b0; d = 8; break;
    case 1:  W = W1; bias = b1; d = 6; break;
    case 2:  W = W2; bias = b2; d = 4; break;
    default: W = W3; bias = b3; d = 8; break;
  }
  float acc = bias[k];
  const float* orow = obs + (size_t)b * OBS_N;
#pragma unroll 8
  for (int a = 0; a < d; ++a)
    acc += orow[obs_idx[n * 8 + a]] * W[a * D_IN + k];
  emb_bf[((size_t)b * NND + n) * D_IN + k] = (bf16_t)tanhf(acc);
}

// ---------------------------------------------------------------------------
// Kernel 4: fused 2-layer GCN, one batch item per workgroup (8 waves), all WMMA.
// ---------------------------------------------------------------------------
__global__ __launch_bounds__(256) void gcn_kernel(
    const bf16_t* __restrict__ emb_bf, const bf16_t* __restrict__ adj_bf,
    const bf16_t* __restrict__ Wg1t, const float* __restrict__ bg1,
    const bf16_t* __restrict__ Wg2t, const float* __restrict__ bg2,
    bf16_t* __restrict__ h2_bf) {
  __shared__ bf16_t bufT[NND * D_H];  // H1t / H2t : [feat][node]  (32 KB)
  __shared__ bf16_t h1s[NND * D_H];   // h1        : [node][feat]  (32 KB)
  int b = blockIdx.x;
  int lane = threadIdx.x & 31, w = threadIdx.x >> 5;
  int half = lane >> 4, l16 = lane & 15;
  const bf16_t* A0 = emb_bf + (size_t)b * (NND * D_IN);

  // Stage 1: H1 = emb_b @ Wg1 -> bufT[n*64+m]
  for (int t = w; t < 64; t += 8) {
    int mt = t >> 4, nt = t & 15;
    const bf16_t* ap = A0 + (mt * 16 + l16) * D_IN;
    const bf16_t* bp = Wg1t + (size_t)(nt * 16 + l16) * D_IN;
    v8f c = {};
#pragma unroll
    for (int kt = 0; kt < D_IN / 32; ++kt)
      c = wmma_bf16(ldfrag(ap, kt * 32, half), ldfrag(bp, kt * 32, half), c);
#pragma unroll
    for (int i = 0; i < 8; ++i)
      bufT[(nt * 16 + l16) * NND + mt * 16 + i + 8 * half] = (bf16_t)c[i];
  }
  __syncthreads();

  // Stage 1b: h1 = tanh(Adj @ H1 + bg1) -> h1s[m*256+n]
  for (int t = w; t < 64; t += 8) {
    int mt = t >> 4, nt = t & 15;
    const bf16_t* ap = adj_bf + (mt * 16 + l16) * NND;
    const bf16_t* bp = bufT + (nt * 16 + l16) * NND;
    v8f c = {};
#pragma unroll
    for (int kt = 0; kt < NND / 32; ++kt)
      c = wmma_bf16(ldfrag(ap, kt * 32, half), ldfrag(bp, kt * 32, half), c);
    float bias = bg1[nt * 16 + l16];
#pragma unroll
    for (int i = 0; i < 8; ++i)
      h1s[(mt * 16 + i + 8 * half) * D_H + nt * 16 + l16] = (bf16_t)tanhf(c[i] + bias);
  }
  __syncthreads();

  // Stage 2: H2 = h1 @ Wg2 -> bufT[n*64+m]
  for (int t = w; t < 64; t += 8) {
    int mt = t >> 4, nt = t & 15;
    const bf16_t* ap = h1s + (mt * 16 + l16) * D_H;
    const bf16_t* bp = Wg2t + (size_t)(nt * 16 + l16) * D_H;
    v8f c = {};
#pragma unroll
    for (int kt = 0; kt < D_H / 32; ++kt)
      c = wmma_bf16(ldfrag(ap, kt * 32, half), ldfrag(bp, kt * 32, half), c);
#pragma unroll
    for (int i = 0; i < 8; ++i)
      bufT[(nt * 16 + l16) * NND + mt * 16 + i + 8 * half] = (bf16_t)c[i];
  }
  __syncthreads();

  // Stage 2b: h2 = tanh(Adj @ H2 + bg2) -> global (B,64,256) bf16
  bf16_t* out = h2_bf + (size_t)b * (NND * D_H);
  for (int t = w; t < 64; t += 8) {
    int mt = t >> 4, nt = t & 15;
    const bf16_t* ap = adj_bf + (mt * 16 + l16) * NND;
    const bf16_t* bp = bufT + (nt * 16 + l16) * NND;
    v8f c = {};
#pragma unroll
    for (int kt = 0; kt < NND / 32; ++kt)
      c = wmma_bf16(ldfrag(ap, kt * 32, half), ldfrag(bp, kt * 32, half), c);
    float bias = bg2[nt * 16 + l16];
#pragma unroll
    for (int i = 0; i < 8; ++i)
      out[(mt * 16 + i + 8 * half) * D_H + nt * 16 + l16] = (bf16_t)tanhf(c[i] + bias);
  }
}

// ---------------------------------------------------------------------------
// Kernel 5: both output heads in one launch (blockIdx.z selects head).
//   out[M,N] = tanh(A[M,K] @ W[K,N] + bias), W stored transposed Wt[n][k] bf16.
// Tile 128(M) x 64(N) per WG, 8 waves x (32x32 each). K staged KC=64 at a time,
// double-buffered:
//   A-tile (128xKC, 16KB): one TDM tensor_load_to_lds per stage (TENSORcnt)
//   B-tile (64xKC,   8KB): per-lane global_load_async_to_lds_b128 (ASYNCcnt)
// Next stage's DMA overlaps current stage's 8 WMMAs per wave.
// ---------------------------------------------------------------------------
__global__ __launch_bounds__(256) void heads_gemm_kernel(
    const bf16_t* __restrict__ A_pi, const bf16_t* __restrict__ Wt_pi,
    const float* __restrict__ bias_pi, float* __restrict__ out_pi,
    const bf16_t* __restrict__ A_vf, const bf16_t* __restrict__ Wt_vf,
    const float* __restrict__ bias_vf, float* __restrict__ out_vf) {
  __shared__ bf16_t Ash[2][128 * KC];  // 2 x 16 KB
  __shared__ bf16_t Bsh[2][64 * KC];   // 2 x  8 KB
  const bf16_t* A; const bf16_t* Wt; const float* bias; float* out; int K;
  if (blockIdx.z == 0) { A = A_pi; Wt = Wt_pi; bias = bias_pi; out = out_pi; K = NND * D_H; }
  else                 { A = A_vf; Wt = Wt_vf; bias = bias_vf; out = out_vf; K = NND * D_IN; }

  int tid = threadIdx.x;
  int lane = tid & 31, w = tid >> 5;
  int half = lane >> 4, l16 = lane & 15;
  int mtile0 = blockIdx.x * 128;
  int ntile0 = blockIdx.y * 64;
  int wm = (w >> 1) * 32;  // wave's 32-row block within tile
  int wn = (w & 1) * 32;   // wave's 32-col block within tile

  // ---- B staging: 64 rows x 128B = 512 x 16B chunks; thread does c=tid, tid+256
  int brow0 = tid >> 3, brow1 = (tid + 256) >> 3, bseg = tid & 7;
  const bf16_t* bSrc0 = Wt + (size_t)(ntile0 + brow0) * K + bseg * 8;
  const bf16_t* bSrc1 = Wt + (size_t)(ntile0 + brow1) * K + bseg * 8;
  unsigned bB0 = ldsaddr(&Bsh[0][0]) + (unsigned)tid * 16u;
  unsigned bB1 = bB0 + 256u * 16u;
  const unsigned bPitch = 64 * KC * 2;

  // ---- A staging
  unsigned aLds = ldsaddr(&Ash[0][0]);
  const unsigned aPitch = 128 * KC * 2;
#if HAVE_TDM
  unsigned long long aG = (unsigned long long)(const void*)(A + (size_t)mtile0 * K);
#else
  const bf16_t* aSrcs[4];
  unsigned aDsts[4];
#pragma unroll
  for (int j = 0; j < 4; ++j) {
    int c = tid + j * 256;
    aSrcs[j] = A + (size_t)(mtile0 + (c >> 3)) * K + (c & 7) * 8;
    aDsts[j] = aLds + (unsigned)c * 16u;
  }
#endif

  int T = K / KC;
  v8f c00 = {}, c01 = {}, c10 = {}, c11 = {};

  // ---- prologue: stage 0 into buffer 0
#if HAVE_TDM
  if (w == 0) tdm_load_tile(aLds, aG, (unsigned)K, 128u);
#else
#pragma unroll
  for (int j = 0; j < 4; ++j) async_copy_b128(aDsts[j], aSrcs[j]);
#endif
  async_copy_b128(bB0, bSrc0);
  async_copy_b128(bB1, bSrc1);

  for (int kt = 0; kt < T; ++kt) {
    int cur = kt & 1;
    if (kt + 1 < T) {
      int nxt = cur ^ 1;
#if HAVE_TDM
      if (w == 0)
        tdm_load_tile(aLds + nxt * aPitch, aG + (unsigned long long)(kt + 1) * (KC * 2),
                      (unsigned)K, 128u);
#else
#pragma unroll
      for (int j = 0; j < 4; ++j)
        async_copy_b128(aDsts[j] + nxt * aPitch, aSrcs[j] + (kt + 1) * KC);
#endif
      async_copy_b128(bB0 + nxt * bPitch, bSrc0 + (kt + 1) * KC);
      async_copy_b128(bB1 + nxt * bPitch, bSrc1 + (kt + 1) * KC);
#if HAVE_TDM
      wait_async_le<2>();                    // current B chunks done
      if (w == 0) wait_tensor_le<1>();       // current A tile done
#else
      wait_async_le<6>();
#endif
    } else {
#if HAVE_TDM
      wait_async_le<0>();
      if (w == 0) __builtin_amdgcn_s_wait_tensorcnt((short)0);
#else
      wait_async_le<0>();
#endif
    }
    __syncthreads();

    const bf16_t* a0p = &Ash[cur][(wm + l16) * KC];
    const bf16_t* a1p = &Ash[cur][(wm + 16 + l16) * KC];
    const bf16_t* b0p = &Bsh[cur][(wn + l16) * KC];
    const bf16_t* b1p = &Bsh[cur][(wn + 16 + l16) * KC];
#pragma unroll
    for (int ks = 0; ks < KC / 32; ++ks) {
      v16bf fa0 = ldfrag(a0p, ks * 32, half);
      v16bf fa1 = ldfrag(a1p, ks * 32, half);
      v16bf fb0 = ldfrag(b0p, ks * 32, half);
      v16bf fb1 = ldfrag(b1p, ks * 32, half);
      c00 = wmma_bf16(fa0, fb0, c00);
      c01 = wmma_bf16(fa0, fb1, c01);
      c10 = wmma_bf16(fa1, fb0, c10);
      c11 = wmma_bf16(fa1, fb1, c11);
    }
    __syncthreads();  // all reads done before buffer is overwritten
  }

  float bb0 = bias[ntile0 + wn + l16];
  float bb1 = bias[ntile0 + wn + 16 + l16];
#pragma unroll
  for (int i = 0; i < 8; ++i) {
    int r0 = mtile0 + wm + i + 8 * half;
    int r1 = r0 + 16;
    out[(size_t)r0 * D_OUT + ntile0 + wn + l16]      = tanhf(c00[i] + bb0);
    out[(size_t)r0 * D_OUT + ntile0 + wn + 16 + l16] = tanhf(c01[i] + bb1);
    out[(size_t)r1 * D_OUT + ntile0 + wn + l16]      = tanhf(c10[i] + bb0);
    out[(size_t)r1 * D_OUT + ntile0 + wn + 16 + l16] = tanhf(c11[i] + bb1);
  }
}

// ---------------------------------------------------------------------------
// Host launcher
// ---------------------------------------------------------------------------
extern "C" void kernel_launch(void* const* d_in, const int* in_sizes, int n_in,
                              void* d_out, int out_size, void* d_ws, size_t ws_size,
                              hipStream_t stream) {
  const float* observations = (const float*)d_in[0];
  const int*   obs_idx      = (const int*)d_in[1];
  const int*   edge_src     = (const int*)d_in[2];
  const int*   edge_dst     = (const int*)d_in[3];
  const float* W_in0 = (const float*)d_in[4];  const float* b_in0 = (const float*)d_in[5];
  const float* W_in1 = (const float*)d_in[6];  const float* b_in1 = (const float*)d_in[7];
  const float* W_in2 = (const float*)d_in[8];  const float* b_in2 = (const float*)d_in[9];
  const float* W_in3 = (const float*)d_in[10]; const float* b_in3 = (const float*)d_in[11];
  const float* Wg1 = (const float*)d_in[12]; const float* bg1 = (const float*)d_in[13];
  const float* Wg2 = (const float*)d_in[14]; const float* bg2 = (const float*)d_in[15];
  const float* Wp  = (const float*)d_in[16]; const float* bp  = (const float*)d_in[17];
  const float* Wv  = (const float*)d_in[18]; const float* bv  = (const float*)d_in[19];
  int nedges = in_sizes[2];

  char* ws = (char*)d_ws;
  size_t off = 0;
  auto carve = [&](size_t bytes) -> char* {
    char* p = ws + off;
    off = (off + bytes + 255) & ~(size_t)255;
    return p;
  };
  bf16_t* adj_bf = (bf16_t*)carve(NND * NND * sizeof(bf16_t));
  bf16_t* Wg1t   = (bf16_t*)carve((size_t)D_IN * D_H * sizeof(bf16_t));
  bf16_t* Wg2t   = (bf16_t*)carve((size_t)D_H * D_H * sizeof(bf16_t));
  bf16_t* Wpt    = (bf16_t*)carve((size_t)(NND * D_H) * D_OUT * sizeof(bf16_t));
  bf16_t* Wvt    = (bf16_t*)carve((size_t)(NND * D_IN) * D_OUT * sizeof(bf16_t));
  bf16_t* emb_bf = (bf16_t*)carve((size_t)B_SZ * NND * D_IN * sizeof(bf16_t));
  bf16_t* h2_bf  = (bf16_t*)carve((size_t)B_SZ * NND * D_H * sizeof(bf16_t));

  float* latent_pi = (float*)d_out;
  float* latent_vf = latent_pi + (size_t)B_SZ * D_OUT;

  prep_kernel<<<1, 512, 0, stream>>>(edge_src, edge_dst, adj_bf, nedges);
  wcvt_kernel<<<(D_IN * D_H + 255) / 256, 256, 0, stream>>>(Wg1, Wg1t, D_IN, D_H);
  wcvt_kernel<<<(D_H * D_H + 255) / 256, 256, 0, stream>>>(Wg2, Wg2t, D_H, D_H);
  wcvt_kernel<<<((NND * D_H) * D_OUT + 255) / 256, 256, 0, stream>>>(Wp, Wpt, NND * D_H, D_OUT);
  wcvt_kernel<<<((NND * D_IN) * D_OUT + 255) / 256, 256, 0, stream>>>(Wv, Wvt, NND * D_IN, D_OUT);
  embed_kernel<<<dim3(B_SZ, NND), D_IN, 0, stream>>>(
      observations, obs_idx, W_in0, b_in0, W_in1, b_in1, W_in2, b_in2, W_in3, b_in3,
      emb_bf);
  gcn_kernel<<<B_SZ, 256, 0, stream>>>(emb_bf, adj_bf, Wg1t, bg1, Wg2t, bg2, h2_bf);
  heads_gemm_kernel<<<dim3(B_SZ / 128, D_OUT / 64, 2), 256, 0, stream>>>(
      h2_bf, Wpt, bp, latent_pi, emb_bf, Wvt, bv, latent_vf);
}
